// Attention_76854144795259
// MI455X (gfx1250) — matrix operand order repaired and checked by reading, not compiled, not run on hardware
//
#include <hip/hip_runtime.h>

// ---------------- CDNA5 WMMA types ----------------
typedef __attribute__((ext_vector_type(16))) __bf16 v16bf;
typedef __attribute__((ext_vector_type(8)))  float  v8f;

#define WMMA_BF16(a, b, c) \
  __builtin_amdgcn_wmma_f32_16x16x32_bf16(false, (a), false, (b), (short)0, (c), false, false)

// B=2, S=2048, D=1024, H=16, HD=64
#define SB 2
#define SS 2048
#define SD 1024
#define SH 16
#define SHD 64
#define SM (SB * SS)      // 4096 rows

// ---- CDNA5 async global->LDS copy (ASYNCcnt-tracked DMA, no VGPR staging) ----
// LDS generic pointers carry the LDS byte offset in their low 32 bits (ISA 10.2:
// LDS aperture truncates to addr[31:0]).
__device__ __forceinline__ void async_copy_b128(void* lds_dst, const void* gsrc) {
  unsigned           la = (unsigned)(unsigned long long)lds_dst;
  unsigned long long ga = (unsigned long long)gsrc;
  asm volatile("global_load_async_to_lds_b128 %0, %1, off"
               :: "v"(la), "v"(ga) : "memory");
}
__device__ __forceinline__ void async_fence() {
  asm volatile("s_wait_asynccnt 0x0" ::: "memory");
}

__device__ __forceinline__ unsigned short f2bf(float f) {
  unsigned int u = __float_as_uint(f);
  u += 0x7fffu + ((u >> 16) & 1u);   // round-to-nearest-even
  return (unsigned short)(u >> 16);
}

__device__ __forceinline__ v16bf frag_cast(uint4 lo, uint4 hi) {
  union { uint4 q[2]; v16bf v; } u;
  u.q[0] = lo; u.q[1] = hi;
  return u.v;
}

// A fragment (16x32 bf16). base -> (row0, k0) of LDS tile, row stride ld (elems).
// Lane l<16: row M=l, K = 0..7 (vgpr0-3) and 16..23 (vgpr4-7); lanes 16-31: K+8.
__device__ __forceinline__ v16bf load_frag_a(const unsigned short* base, int ld) {
  int lane = threadIdx.x & 31;
  int lh = lane >> 4, r = lane & 15;
  const unsigned short* p = base + r * ld + lh * 8;
  uint4 lo = *(const uint4*)(p);
  uint4 hi = *(const uint4*)(p + 16);
  return frag_cast(lo, hi);
}

// B fragment (32x16 bf16) from a tile stored [n][k] (k contiguous).
// Lane: N = lane&15, K = 0..7 and 16..23 for lanes 0-15; lanes 16-31: K+8.
__device__ __forceinline__ v16bf load_frag_b(const unsigned short* base, int ld) {
  int lane = threadIdx.x & 31;
  int lh = lane >> 4, r = lane & 15;
  const unsigned short* p = base + r * ld + lh * 8;
  uint4 lo = *(const uint4*)(p);
  uint4 hi = *(const uint4*)(p + 16);
  return frag_cast(lo, hi);
}

// ---------------- fp32 -> bf16 conversion ----------------
__global__ void convert_f32_bf16(const float* __restrict__ in,
                                 unsigned short* __restrict__ out, int n4) {
  int i = blockIdx.x * blockDim.x + threadIdx.x;
  if (i < n4) {
    float4 f = ((const float4*)in)[i];
    unsigned long long v =  (unsigned long long)f2bf(f.x)
                         | ((unsigned long long)f2bf(f.y) << 16)
                         | ((unsigned long long)f2bf(f.z) << 32)
                         | ((unsigned long long)f2bf(f.w) << 48);
    ((unsigned long long*)out)[i] = v;
  }
}

// ---------------- fused QKV projection GEMM ----------------
// y = x @ W^T ; A[m][k] (bf16), W[n][k] (bf16). 64x64 tile, 4 waves, K-step 32.
// Output scattered to [b,h,s,hd] bf16.
#define LDT 40   // 32 + 8 pad (16B-aligned rows: 80B)

__global__ __launch_bounds__(128) void gemm_qkv(
    const unsigned short* __restrict__ A,
    const unsigned short* __restrict__ Wq,
    const unsigned short* __restrict__ Wk,
    const unsigned short* __restrict__ Wv,
    unsigned short* __restrict__ Qo,
    unsigned short* __restrict__ Ko,
    unsigned short* __restrict__ Vo) {
  __shared__ unsigned short As[64 * LDT];
  __shared__ unsigned short Bs[64 * LDT];

  const unsigned short* W = (blockIdx.z == 0) ? Wq : ((blockIdx.z == 1) ? Wk : Wv);
  unsigned short* O = (blockIdx.z == 0) ? Qo : ((blockIdx.z == 1) ? Ko : Vo);

  int m0 = blockIdx.x * 64, n0 = blockIdx.y * 64;
  int t = threadIdx.x;
  int wave = t >> 5, lane = t & 31, lh = lane >> 4, r = lane & 15;

  v8f acc[4] = {};

  for (int k0 = 0; k0 < SD; k0 += 32) {
    __syncthreads();
    for (int c = t; c < 256; c += 128) {   // 64 rows x 32 cols, 8 elems/chunk
      int row = c >> 2, cc = (c & 3) * 8;
      async_copy_b128(&As[row * LDT + cc], &A[(m0 + row) * SD + k0 + cc]);
      async_copy_b128(&Bs[row * LDT + cc], &W[(n0 + row) * SD + k0 + cc]);
    }
    async_fence();
    __syncthreads();
    v16bf af = load_frag_a(&As[(wave * 16) * LDT], LDT);
#pragma unroll
    for (int f = 0; f < 4; ++f) {
      v16bf bf = load_frag_b(&Bs[(f * 16) * LDT], LDT);
      acc[f] = WMMA_BF16(af, bf, acc[f]);
    }
  }
#pragma unroll
  for (int f = 0; f < 4; ++f) {
#pragma unroll
    for (int v = 0; v < 8; ++v) {
      int m = m0 + wave * 16 + v + 8 * lh;
      int n = n0 + f * 16 + r;
      int bb = m >> 11, s = m & (SS - 1);
      int h = n >> 6, hd = n & (SHD - 1);
      O[(((bb << 4) + h) * SS + s) * SHD + hd] = f2bf(acc[f][v]);
    }
  }
}

// ---------------- flash attention ----------------
// grid: (S/64 q-blocks, B*H). block = 128 (4 waves); each wave owns 16 q rows.
#define LDA 72   // 64 + 8 pad (144B rows, 16B aligned)

__global__ __launch_bounds__(128) void attn_kernel(
    const unsigned short* __restrict__ Q,   // [B*H][S][HD]
    const unsigned short* __restrict__ Kk,
    const unsigned short* __restrict__ V,
    unsigned short* __restrict__ Y) {       // [B*S][D], col = h*64+hd
  __shared__ unsigned short Qs[64 * LDA];
  __shared__ unsigned short Ks[64 * LDA];
  __shared__ unsigned short Vt[64 * LDA];   // transposed: [hd][seq]
  __shared__ unsigned short Ps[64 * LDA];   // wave w owns rows w*16..w*16+15

  int qb = blockIdx.x;
  int bh = blockIdx.y;
  int q0 = qb * 64;
  const unsigned short* Qb = Q + (size_t)bh * SS * SHD;
  const unsigned short* Kb = Kk + (size_t)bh * SS * SHD;
  const unsigned short* Vb = V + (size_t)bh * SS * SHD;

  int t = threadIdx.x, wave = t >> 5, lane = t & 31, lh = lane >> 4, r = lane & 15;

  for (int c = t; c < 512; c += 128) {      // Q tile 64x64 (async DMA)
    int row = c >> 3, cc = (c & 7) * 8;
    async_copy_b128(&Qs[row * LDA + cc], &Qb[(q0 + row) * SHD + cc]);
  }

  v8f accO[4] = {};
  float mrow[8], lrow[8];
#pragma unroll
  for (int v = 0; v < 8; ++v) { mrow[v] = -1e30f; lrow[v] = 0.f; }

  for (int kb = 0; kb <= qb; ++kb) {
    int kk0 = kb * 64;
    __syncthreads();
    for (int c = t; c < 512; c += 128) {
      int row = c >> 3, cc = (c & 7) * 8;
      async_copy_b128(&Ks[row * LDA + cc], &Kb[(kk0 + row) * SHD + cc]);
      uint4 vv = *(const uint4*)(&Vb[(kk0 + row) * SHD + cc]);
      const unsigned short* pv = (const unsigned short*)&vv;
#pragma unroll
      for (int e = 0; e < 8; ++e) Vt[(cc + e) * LDA + row] = pv[e];  // transpose
    }
    async_fence();
    __syncthreads();

    // S = Q K^T  (8 WMMAs)
    v8f sfr[4];
#pragma unroll
    for (int f = 0; f < 4; ++f) {
      v8f z = {};
      sfr[f] = z;
#pragma unroll
      for (int kk = 0; kk < 64; kk += 32) {
        v16bf af = load_frag_a(&Qs[(wave * 16) * LDA + kk], LDA);
        v16bf bf = load_frag_b(&Ks[(f * 16) * LDA + kk], LDA);
        sfr[f] = WMMA_BF16(af, bf, sfr[f]);
      }
    }
    // scale + causal mask
    float sv[4][8];
#pragma unroll
    for (int f = 0; f < 4; ++f) {
#pragma unroll
      for (int v = 0; v < 8; ++v) {
        int qg = q0 + wave * 16 + v + 8 * lh;
        int kg = kk0 + f * 16 + r;
        float x = sfr[f][v] * 0.125f;   // 1/sqrt(64)
        sv[f][v] = (kg <= qg) ? x : -1e30f;
      }
    }
    // online softmax: row lives in 16 lanes of one half-wave at fixed vgpr v
#pragma unroll
    for (int v = 0; v < 8; ++v) {
      float mv = fmaxf(fmaxf(sv[0][v], sv[1][v]), fmaxf(sv[2][v], sv[3][v]));
#pragma unroll
      for (int off = 1; off < 16; off <<= 1) mv = fmaxf(mv, __shfl_xor(mv, off, 32));
      float mnew = fmaxf(mrow[v], mv);
      float alpha = __expf(mrow[v] - mnew);
      mrow[v] = mnew;
      float rs = 0.f;
#pragma unroll
      for (int f = 0; f < 4; ++f) {
        float pe = __expf(sv[f][v] - mnew);
        sv[f][v] = pe;
        rs += pe;
      }
#pragma unroll
      for (int off = 1; off < 16; off <<= 1) rs += __shfl_xor(rs, off, 32);
      lrow[v] = lrow[v] * alpha + rs;
#pragma unroll
      for (int f = 0; f < 4; ++f) accO[f][v] *= alpha;
    }
    // C-layout -> A-layout via LDS (bf16 P)
#pragma unroll
    for (int f = 0; f < 4; ++f)
#pragma unroll
      for (int v = 0; v < 8; ++v)
        Ps[(wave * 16 + v + 8 * lh) * LDA + f * 16 + r] = f2bf(sv[f][v]);
    __syncthreads();
    // O += P V  (8 WMMAs)
#pragma unroll
    for (int f = 0; f < 4; ++f) {
#pragma unroll
      for (int kk = 0; kk < 64; kk += 32) {
        v16bf af = load_frag_a(&Ps[(wave * 16) * LDA + kk], LDA);
        v16bf bf = load_frag_b(&Vt[(f * 16) * LDA + kk], LDA);
        accO[f] = WMMA_BF16(af, bf, accO[f]);
      }
    }
  }
  // normalize + store to [b, s, h*64+hd] bf16
  int h = bh & (SH - 1), bb = bh >> 4;
#pragma unroll
  for (int f = 0; f < 4; ++f) {
#pragma unroll
    for (int v = 0; v < 8; ++v) {
      int sq = q0 + wave * 16 + v + 8 * lh;
      float val = accO[f][v] / lrow[v];
      Y[((size_t)bb * SS + sq) * SD + h * SHD + f * 16 + r] = f2bf(val);
    }
  }
}

// ---------------- output projection GEMM (f32 out) ----------------
__global__ __launch_bounds__(128) void gemm_out(
    const unsigned short* __restrict__ A,   // y_att [4096 x 1024] bf16
    const unsigned short* __restrict__ W,   // wo [1024 x 1024] bf16
    float* __restrict__ O) {
  __shared__ unsigned short As[64 * LDT];
  __shared__ unsigned short Bs[64 * LDT];

  int m0 = blockIdx.x * 64, n0 = blockIdx.y * 64;
  int t = threadIdx.x;
  int wave = t >> 5, lane = t & 31, lh = lane >> 4, r = lane & 15;

  v8f acc[4] = {};

  for (int k0 = 0; k0 < SD; k0 += 32) {
    __syncthreads();
    for (int c = t; c < 256; c += 128) {
      int row = c >> 2, cc = (c & 3) * 8;
      async_copy_b128(&As[row * LDT + cc], &A[(m0 + row) * SD + k0 + cc]);
      async_copy_b128(&Bs[row * LDT + cc], &W[(n0 + row) * SD + k0 + cc]);
    }
    async_fence();
    __syncthreads();
    v16bf af = load_frag_a(&As[(wave * 16) * LDT], LDT);
#pragma unroll
    for (int f = 0; f < 4; ++f) {
      v16bf bf = load_frag_b(&Bs[(f * 16) * LDT], LDT);
      acc[f] = WMMA_BF16(af, bf, acc[f]);
    }
  }
#pragma unroll
  for (int f = 0; f < 4; ++f) {
#pragma unroll
    for (int v = 0; v < 8; ++v) {
      int m = m0 + wave * 16 + v + 8 * lh;
      int n = n0 + f * 16 + r;
      O[(size_t)m * SD + n] = acc[f][v];
    }
  }
}

// ---------------- host launch ----------------
extern "C" void kernel_launch(void* const* d_in, const int* in_sizes, int n_in,
                              void* d_out, int out_size, void* d_ws, size_t ws_size,
                              hipStream_t stream) {
  const float* x  = (const float*)d_in[0];
  const float* wq = (const float*)d_in[1];
  const float* wk = (const float*)d_in[2];
  const float* wv = (const float*)d_in[3];
  const float* wo = (const float*)d_in[4];

  unsigned short* ws  = (unsigned short*)d_ws;
  unsigned short* xb  = ws;                         // 4096*1024
  unsigned short* wqb = xb  + (size_t)SM * SD;
  unsigned short* wkb = wqb + (size_t)SD * SD;
  unsigned short* wvb = wkb + (size_t)SD * SD;
  unsigned short* wob = wvb + (size_t)SD * SD;
  unsigned short* qb  = wob + (size_t)SD * SD;      // [32][2048][64]
  unsigned short* kb  = qb  + (size_t)SM * SD;
  unsigned short* vb  = kb  + (size_t)SM * SD;
  unsigned short* ya  = vb  + (size_t)SM * SD;

  const int nx = SM * SD, nw = SD * SD;
  convert_f32_bf16<<<(nx / 4 + 255) / 256, 256, 0, stream>>>(x,  xb,  nx / 4);
  convert_f32_bf16<<<(nw / 4 + 255) / 256, 256, 0, stream>>>(wq, wqb, nw / 4);
  convert_f32_bf16<<<(nw / 4 + 255) / 256, 256, 0, stream>>>(wk, wkb, nw / 4);
  convert_f32_bf16<<<(nw / 4 + 255) / 256, 256, 0, stream>>>(wv, wvb, nw / 4);
  convert_f32_bf16<<<(nw / 4 + 255) / 256, 256, 0, stream>>>(wo, wob, nw / 4);

  dim3 g1(SM / 64, SD / 64, 3);
  gemm_qkv<<<g1, 128, 0, stream>>>(xb, wqb, wkb, wvb, qb, kb, vb);

  dim3 g2(SS / 64, SB * SH, 1);
  attn_kernel<<<g2, 128, 0, stream>>>(qb, kb, vb, ya);

  dim3 g3(SM / 64, SD / 64, 1);
  gemm_out<<<g3, 128, 0, stream>>>(ya, wob, (float*)d_out);
}
